// TrfEdgeNet_76321568850284
// MI455X (gfx1250) — matrix operand order repaired and checked by reading, not compiled, not run on hardware
//
#include <hip/hip_runtime.h>
#include <math.h>

// ---------------------------------------------------------------------------
// Types
// ---------------------------------------------------------------------------
typedef __bf16 bf16_t;
typedef __attribute__((ext_vector_type(16))) __bf16   v16bf;
typedef __attribute__((ext_vector_type(8)))  float    v8f;
typedef __attribute__((ext_vector_type(4)))  float    f32x4;
typedef __attribute__((ext_vector_type(2)))  unsigned u32x2;
typedef __attribute__((ext_vector_type(4)))  unsigned u32x4;
typedef __attribute__((ext_vector_type(8)))  unsigned u32x8;

// ---------------------------------------------------------------------------
// Device helpers
// ---------------------------------------------------------------------------
__device__ inline unsigned f2bfbits(float f) {            // RNE fp32 -> bf16 bits
    unsigned u = __float_as_uint(f);
    return (u + 0x7FFFu + ((u >> 16) & 1u)) >> 16;
}
__device__ inline bf16_t f2bf(float f) {
    return __builtin_bit_cast(bf16_t, (unsigned short)f2bfbits(f));
}
// monotone float <-> uint mapping so atomicMax(uint) == float max
__device__ inline unsigned f2ord(float f) {
    unsigned u = __float_as_uint(f);
    return (u & 0x80000000u) ? ~u : (u | 0x80000000u);
}
__device__ inline float ord2f(unsigned u) {
    unsigned v = (u & 0x80000000u) ? (u & 0x7FFFFFFFu) : ~u;
    return __uint_as_float(v);
}
// two 16B loads -> one 32B bf16 fragment (kept in VGPRs)
__device__ inline v16bf load_frag(const bf16_t* p0, const bf16_t* p1) {
    u32x4 a = *(const u32x4*)p0;
    u32x4 b = *(const u32x4*)p1;
    u32x8 t;
    t[0] = a[0]; t[1] = a[1]; t[2] = a[2]; t[3] = a[3];
    t[4] = b[0]; t[5] = b[1]; t[6] = b[2]; t[7] = b[3];
    return __builtin_bit_cast(v16bf, t);
}

// ---------------------------------------------------------------------------
// Weight prep: W[K][Nn] fp32 (row-major) -> WT[Nn][K] bf16 (row-major),
// LDS-tiled transpose so both global streams stay coalesced.
// ---------------------------------------------------------------------------
__global__ void transpose_bf16_kernel(const float* __restrict__ W,
                                      bf16_t* __restrict__ WT, int K, int Nn) {
    __shared__ float tile[32][33];
    int k0 = blockIdx.x * 32;
    int n0 = blockIdx.y * 32;
    int tx = threadIdx.x;        // 0..31
    int ty = threadIdx.y;        // 0..7
    for (int i = ty; i < 32; i += 8) {
        int k = k0 + i, n = n0 + tx;
        tile[i][tx] = (k < K && n < Nn) ? W[(size_t)k * Nn + n] : 0.0f;
    }
    __syncthreads();
    for (int i = ty; i < 32; i += 8) {
        int n = n0 + i, k = k0 + tx;
        if (n < Nn && k < K) WT[(size_t)n * K + k] = f2bf(tile[tx][i]);
    }
}

// fp32 -> bf16 (optionally fused ReLU), n must be a multiple of 4 (it is here)
__global__ void convert_bf16_kernel(const float* __restrict__ s,
                                    bf16_t* __restrict__ d, long long n, int relu) {
    long long i = ((long long)blockIdx.x * blockDim.x + threadIdx.x) * 4;
    if (i >= n) return;
    f32x4 v = *(const f32x4*)(s + i);
    if (relu) {
        v[0] = fmaxf(v[0], 0.f); v[1] = fmaxf(v[1], 0.f);
        v[2] = fmaxf(v[2], 0.f); v[3] = fmaxf(v[3], 0.f);
    }
    u32x2 o;
    o[0] = f2bfbits(v[0]) | (f2bfbits(v[1]) << 16);
    o[1] = f2bfbits(v[2]) | (f2bfbits(v[3]) << 16);
    *(u32x2*)(d + i) = o;
}

__global__ void fill_u32_kernel(unsigned* __restrict__ p, unsigned v, long long n) {
    long long i = (long long)blockIdx.x * blockDim.x + threadIdx.x;
    if (i < n) p[i] = v;
}

// ---------------------------------------------------------------------------
// WMMA GEMM: D[M][Nn] = A[M][KK](bf16) * WT[Nn][KK](bf16)^T (+ bias)
// NT and KK are compile-time: the whole MAC body is one flat SSA DAG of
// global_load_b128 (immediate offsets rooted at the kernel args -> provably
// addrspace(1), no flat path) + v_wmma, so the scheduler can hoist loads far
// ahead of their consuming WMMA (partial s_wait_loadcnt instead of wait-0).
// wave = 16 rows x (NT*16) cols; block = 128 threads = 4 waves.
// NT=4: requires Nn % 64 == 0 (all projection widths). NT=1: classifier.
// Fragment layouts per CDNA5 ISA 16-bit A(16x32)/B(32x16):
//  A: lane<16 -> row=lane, K {k0..k0+7, k0+16..k0+23};
//     lane>=16 -> row=lane-16, K {k0+8..k0+15, k0+24..k0+31}
//  B: lane<16 -> col=lane, K {k0..k0+7, k0+16..k0+23};
//     lane>=16 -> col=lane-16, K {k0+8..k0+15, k0+24..k0+31}
//  C/D: vgpr i, lane<16 -> (M=i, N=lane); lane>=16 -> (M=8+i, N=lane-16)
// ---------------------------------------------------------------------------
template <bool OUT_BF16, int NT, int KK>
__global__ __launch_bounds__(128)
void gemm_wmma_kernel(const bf16_t* __restrict__ A, const bf16_t* __restrict__ WT,
                      const float* __restrict__ bias, void* __restrict__ Dv,
                      int M, int Nn) {
    const int wave = threadIdx.x >> 5;
    const int lane = threadIdx.x & 31;
    const int row0 = blockIdx.x << 4;
    const int col0 = blockIdx.y * (NT * 64) + wave * (NT * 16);
    if (col0 >= Nn) return;                       // wave-uniform: EXEC stays full

    const int lr = lane & 15;
    const int hi = lane >> 4;
    const size_t abase = (size_t)(row0 + lr) * KK + hi * 8;   // elements
    const size_t bbase = (size_t)(col0 + lr) * KK + hi * 8;   // elements

    v8f acc[NT];
#pragma unroll
    for (int t = 0; t < NT; ++t)
        acc[t] = v8f{0.f, 0.f, 0.f, 0.f, 0.f, 0.f, 0.f, 0.f};

#pragma unroll
    for (int k0 = 0; k0 < KK; k0 += 32) {
        v16bf af = load_frag(A + abase + k0, A + abase + k0 + 16);
#pragma unroll
        for (int t = 0; t < NT; ++t) {
            const bf16_t* bp = WT + bbase + (size_t)t * (16 * KK) + k0;
            v16bf bfr = load_frag(bp, bp + 16);
            acc[t] = __builtin_amdgcn_wmma_f32_16x16x32_bf16(
                false, af, false, bfr, (short)0, acc[t], false, false);
        }
    }
#pragma unroll
    for (int t = 0; t < NT; ++t) {
        int col = col0 + t * 16 + lr;
        float bv = bias ? bias[col] : 0.0f;
#pragma unroll
        for (int i = 0; i < 8; ++i) {
            int r = row0 + (hi << 3) + i;
            float val = acc[t][i] + bv;
            size_t off = (size_t)r * Nn + col;
            if (OUT_BF16) ((bf16_t*)Dv)[off] = f2bf(val);
            else          ((float*)Dv)[off]  = val;
        }
    }
}

// ---------------------------------------------------------------------------
// Edge phase. wave per (edge, head); H = 8.
// ---------------------------------------------------------------------------
__global__ __launch_bounds__(256)
void edge_alpha_kernel(const float* __restrict__ q, const float* __restrict__ k,
                       const bf16_t* __restrict__ e, const int* __restrict__ src,
                       const int* __restrict__ tgt, float* __restrict__ alpha,
                       unsigned* __restrict__ mOrd, int E, int C, float scale) {
    int w = (int)(blockIdx.x * (blockDim.x >> 5)) + (int)(threadIdx.x >> 5);
    int lane = threadIdx.x & 31;
    if (w >= E * 8) return;
    int eid = w >> 3, h = w & 7;
    int s = src[eid], t = tgt[eid];
    size_t HC = (size_t)C * 8;
    const float*  qp = q + (size_t)t * HC + (size_t)h * C;
    const float*  kp = k + (size_t)s * HC + (size_t)h * C;
    const bf16_t* ep = e + (size_t)eid * HC + (size_t)h * C;
    float dot = 0.f;
    for (int c = lane * 4; c < C; c += 128) {
        f32x4 qv = *(const f32x4*)(qp + c);
        f32x4 kv = *(const f32x4*)(kp + c);
        u32x2 eu = *(const u32x2*)(ep + c);
        float e0 = __uint_as_float(eu[0] << 16);
        float e1 = __uint_as_float(eu[0] & 0xFFFF0000u);
        float e2 = __uint_as_float(eu[1] << 16);
        float e3 = __uint_as_float(eu[1] & 0xFFFF0000u);
        dot += qv[0] * (kv[0] + e0) + qv[1] * (kv[1] + e1)
             + qv[2] * (kv[2] + e2) + qv[3] * (kv[3] + e3);
    }
    for (int o = 16; o > 0; o >>= 1) dot += __shfl_xor(dot, o, 32);
    dot *= scale;
    if (lane == 0) {
        alpha[w] = dot;
        atomicMax(&mOrd[(size_t)t * 8 + h], f2ord(dot));
    }
}

__global__ void edge_expsum_kernel(float* __restrict__ alpha,
                                   const unsigned* __restrict__ mOrd,
                                   const int* __restrict__ tgt,
                                   float* __restrict__ den, int E) {
    int w = blockIdx.x * blockDim.x + threadIdx.x;
    if (w >= E * 8) return;
    int eid = w >> 3, h = w & 7;
    int t = tgt[eid];
    float m = ord2f(mOrd[(size_t)t * 8 + h]);
    float ex = __expf(alpha[w] - m);
    alpha[w] = ex;
    atomicAdd(&den[(size_t)t * 8 + h], ex);
}

__global__ __launch_bounds__(256)
void edge_scatter_kernel(const float* __restrict__ v, const bf16_t* __restrict__ e,
                         const float* __restrict__ alpha, const float* __restrict__ den,
                         const int* __restrict__ src, const int* __restrict__ tgt,
                         float* __restrict__ hout, int E, int C) {
    int w = (int)(blockIdx.x * (blockDim.x >> 5)) + (int)(threadIdx.x >> 5);
    int lane = threadIdx.x & 31;
    if (w >= E * 8) return;
    int eid = w >> 3, h = w & 7;
    int s = src[eid], t = tgt[eid];
    float a = alpha[w] / (den[(size_t)t * 8 + h] + 1e-16f) * 0.125f; // /H for head-mean
    size_t HC = (size_t)C * 8;
    const float*  vp = v + (size_t)s * HC + (size_t)h * C;
    const bf16_t* ep = e + (size_t)eid * HC + (size_t)h * C;
    float* op = hout + (size_t)t * C;
    for (int c = lane * 4; c < C; c += 128) {
        f32x4 vv = *(const f32x4*)(vp + c);
        u32x2 eu = *(const u32x2*)(ep + c);
        float e0 = __uint_as_float(eu[0] << 16);
        float e1 = __uint_as_float(eu[0] & 0xFFFF0000u);
        float e2 = __uint_as_float(eu[1] << 16);
        float e3 = __uint_as_float(eu[1] & 0xFFFF0000u);
        atomicAdd(op + c + 0, a * (vv[0] + e0));
        atomicAdd(op + c + 1, a * (vv[1] + e1));
        atomicAdd(op + c + 2, a * (vv[2] + e2));
        atomicAdd(op + c + 3, a * (vv[3] + e3));
    }
}

// ---------------------------------------------------------------------------
// Host orchestration
// ---------------------------------------------------------------------------
static inline int ceil_div(int a, int b) { return (a + b - 1) / b; }

template <bool OB, int NT, int KK>
static void launch_gemm(const bf16_t* A, const bf16_t* T, const float* bias,
                        void* D, int M, int Nn, hipStream_t stream) {
    dim3 g(M / 16, ceil_div(Nn, NT * 64));
    gemm_wmma_kernel<OB, NT, KK><<<g, 128, 0, stream>>>(A, T, bias, D, M, Nn);
}

extern "C" void kernel_launch(void* const* d_in, const int* in_sizes, int n_in,
                              void* d_out, int out_size, void* d_ws, size_t ws_size,
                              hipStream_t stream) {
    (void)in_sizes; (void)n_in; (void)out_size; (void)ws_size;
    const int N = 10000, E = 160000, H = 8;

    const float* x  = (const float*)d_in[0];
    const int*   ei = (const int*)d_in[1];        // [2,E] int32 (jax x64 disabled)
    const float* ea = (const float*)d_in[2];
    const int* srcp = ei;
    const int* tgtp = ei + E;

    struct P { const float *Wq,*bq,*Wk,*bk,*Wv,*bv,*We,*Ws,*bs; };
    P L[3];
    int fins[3] = {128, 128, 128};
    int cs[3]   = {128, 128, 256};
    int base = 3;
    for (int l = 0; l < 3; ++l) {
        L[l].Wq = (const float*)d_in[base + 0]; L[l].bq = (const float*)d_in[base + 1];
        L[l].Wk = (const float*)d_in[base + 2]; L[l].bk = (const float*)d_in[base + 3];
        L[l].Wv = (const float*)d_in[base + 4]; L[l].bv = (const float*)d_in[base + 5];
        L[l].We = (const float*)d_in[base + 6];
        L[l].Ws = (const float*)d_in[base + 7]; L[l].bs = (const float*)d_in[base + 8];
        base += 9;
    }
    const float* Wc = (const float*)d_in[30];
    const float* bc = (const float*)d_in[31];

    // --- workspace bump allocator (total ~0.96 GB; e-buffer bf16 dominates) ---
    char* wsp = (char*)d_ws;
    auto alloc = [&](size_t bytes) -> void* {
        void* p = (void*)wsp;
        wsp += (bytes + 255) & ~(size_t)255;
        return p;
    };
    bf16_t*   xb   = (bf16_t*)alloc((size_t)N * 256 * 2);     // current activations bf16
    bf16_t*   eb   = (bf16_t*)alloc((size_t)E * 64 * 2);      // edge_attr bf16
    float*    qb   = (float*) alloc((size_t)N * 2048 * 4);
    float*    kb   = (float*) alloc((size_t)N * 2048 * 4);
    float*    vb   = (float*) alloc((size_t)N * 2048 * 4);
    bf16_t*   ebuf = (bf16_t*)alloc((size_t)E * 2048 * 2);    // edge embedding bf16
    float*    alph = (float*) alloc((size_t)E * H * 4);
    unsigned* mOrd = (unsigned*)alloc((size_t)N * H * 4);
    float*    den  = (float*) alloc((size_t)N * H * 4);
    float*    hout = (float*) alloc((size_t)N * 256 * 4);
    bf16_t*   WT[3][5];                                        // WqT,WkT,WvT,WeT,WsT
    for (int l = 0; l < 3; ++l) {
        int HC = H * cs[l];
        WT[l][0] = (bf16_t*)alloc((size_t)HC * fins[l] * 2);
        WT[l][1] = (bf16_t*)alloc((size_t)HC * fins[l] * 2);
        WT[l][2] = (bf16_t*)alloc((size_t)HC * fins[l] * 2);
        WT[l][3] = (bf16_t*)alloc((size_t)HC * 64 * 2);
        WT[l][4] = (bf16_t*)alloc((size_t)cs[l] * fins[l] * 2);
    }
    bf16_t* WcT = (bf16_t*)alloc((size_t)16 * 256 * 2);

    auto transpose = [&](const float* W, bf16_t* T, int K, int Nn) {
        dim3 g(ceil_div(K, 32), ceil_div(Nn, 32)), b(32, 8);
        transpose_bf16_kernel<<<g, b, 0, stream>>>(W, T, K, Nn);
    };
    auto convert = [&](const float* s, bf16_t* d, long long n, int relu) {
        int grid = (int)((n / 4 + 255) / 256);
        convert_bf16_kernel<<<grid, 256, 0, stream>>>(s, d, n, relu);
    };
    auto fill = [&](unsigned* p, unsigned v, long long n) {
        fill_u32_kernel<<<(int)((n + 255) / 256), 256, 0, stream>>>(p, v, n);
    };

    // --- weight prep (small; re-run every call for determinism) ---
    for (int l = 0; l < 3; ++l) {
        int HC = H * cs[l];
        transpose(L[l].Wq, WT[l][0], fins[l], HC);
        transpose(L[l].Wk, WT[l][1], fins[l], HC);
        transpose(L[l].Wv, WT[l][2], fins[l], HC);
        transpose(L[l].We, WT[l][3], 64, HC);
        transpose(L[l].Ws, WT[l][4], fins[l], cs[l]);
    }
    transpose(Wc, WcT, 256, 16);
    convert(ea, eb, (long long)E * 64, 0);
    convert(x,  xb, (long long)N * 128, 0);

    const unsigned ORD_NEG_INF = 0x007FFFFFu;  // f2ord(-inf)
    for (int l = 0; l < 3; ++l) {
        int C = cs[l], HC = H * C;
        // Fin == 128 for all three layers (D1 == F_NODE == 128)
        launch_gemm<false, 4, 128>(xb, WT[l][0], L[l].bq, qb, N, HC, stream);
        launch_gemm<false, 4, 128>(xb, WT[l][1], L[l].bk, kb, N, HC, stream);
        launch_gemm<false, 4, 128>(xb, WT[l][2], L[l].bv, vb, N, HC, stream);
        launch_gemm<true,  4, 64 >(eb, WT[l][3], nullptr, ebuf, E, HC, stream);
        launch_gemm<false, 4, 128>(xb, WT[l][4], L[l].bs, hout, N, C, stream); // skip seeds hout
        fill(mOrd, ORD_NEG_INF, (long long)N * H);
        fill((unsigned*)den, 0u, (long long)N * H);

        float scale = 1.0f / sqrtf((float)C);
        int nw = E * H;                       // one wave per (edge, head)
        int blocks = ceil_div(nw, 8);         // 8 waves / 256-thread block
        edge_alpha_kernel<<<blocks, 256, 0, stream>>>(qb, kb, ebuf, srcp, tgtp,
                                                      alph, mOrd, E, C, scale);
        edge_expsum_kernel<<<ceil_div(nw, 256), 256, 0, stream>>>(alph, mOrd, tgtp,
                                                                  den, E);
        edge_scatter_kernel<<<blocks, 256, 0, stream>>>(vb, ebuf, alph, den,
                                                        srcp, tgtp, hout, E, C);
        // next layer input (ReLU after layers 1,2; none after layer 3)
        convert(hout, xb, (long long)N * C, (l < 2) ? 1 : 0);
    }
    // classifier: [N,256] x [256,16] -> d_out fp32
    launch_gemm<false, 1, 256>(xb, WcT, bc, d_out, N, 16, stream);
}
